// PCFG_91044716741118
// MI455X (gfx1250) — matrix operand order repaired and checked by reading, not compiled, not run on hardware
//
#include <hip/hip_runtime.h>

// ---------------------------------------------------------------------------
// PCFG inside algorithm (neg logZ) for gfx1250 (MI455X, wave32, WMMA).
// B=16, N=30, NT=PT=128, S=256.
//
// Dominant compute: per width step, zm[b,n,a] = sum_k eY[b,n,k]*er[b,a,k]
// with K=65536 -> mapped onto v_wmma_f32_16x16x32_bf16 (bf16 in, f32 acc).
// er (exp(rules-max)) is precomputed once in bf16 (256 MB) and streamed per
// step; MI455X's 192MB L2 keeps most of it resident across the 29 steps.
// The split-combination mm[s,t] = sum_u el[u,s]*eR[u,t] (K<=29, padded to 32)
// is also one WMMA per 16x16 tile.
// ---------------------------------------------------------------------------

#define NEGV  (-1000000000.0f)
#define TINYV (1e-38f)

typedef __attribute__((ext_vector_type(16))) __bf16 v16bf;
typedef __attribute__((ext_vector_type(8)))  float  v8f;

union BF16Frag { v16bf v; unsigned short u[16]; };

static __device__ inline unsigned short f2bf(float f) {
  union { float f; unsigned u; } cv; cv.f = f;
  unsigned u = cv.u;
  u += 0x7fffu + ((u >> 16) & 1u);        // round-to-nearest-even
  return (unsigned short)(u >> 16);
}

// A-matrix 16x32 bf16 element->K map (ISA 7.12.2): e=0..7 -> K=hi8+e,
// e=8..15 -> K=16+hi8+(e-8)... i.e. 16 + hi8 + (e&7).
static __device__ inline int kmapA(int e, int hi) {
  return ((e >> 3) << 4) + (hi ? 8 : 0) + (e & 7);
}
// B-matrix 32x16 bf16: same K map as A (e=0..7 -> hi8+e, e=8..15 -> 16+hi8+(e&7)).
static __device__ inline int kmapB(int e, int hi) {
  return ((e >> 3) << 4) + (hi ? 8 : 0) + (e & 7);
}

// ---------------------------------------------------------------------------
// chart init: chart[b][n][w][s] = NEG, except w==0, s>=NT : terms
// ---------------------------------------------------------------------------
__global__ __launch_bounds__(256) void kInit(float* __restrict__ chart,
                                             const float* __restrict__ terms) {
  const int N = 30, NT = 128, PT = 128;
  int idx = blockIdx.x * 256 + threadIdx.x;
  if (idx >= 16 * N * N * 256) return;
  int s = idx & 255;
  int rest = idx >> 8;
  int uw = rest % N; rest /= N;
  int n  = rest % N;
  int b  = rest / N;
  float v = NEGV;
  if (uw == 0 && s >= NT) v = terms[(b * N + n) * PT + (s - NT)];
  chart[idx] = v;
}

// ---------------------------------------------------------------------------
// rules prep: mru[b][a] = max_k rules, er[b][a][t*256+s] = bf16(exp(r - mru))
// (k-transposed layout so kA's eY stores are contiguous; both GEMM operands
// use the same flat-k permutation, so the k-sum is unchanged.)
// ---------------------------------------------------------------------------
__global__ __launch_bounds__(256) void kRules(const float* __restrict__ rules,
                                              unsigned short* __restrict__ er,
                                              float* __restrict__ mruB) {
  int bid = blockIdx.x;                     // b*128 + a
  const float* base = rules + ((size_t)bid << 16);
  __shared__ float red[256];
  int tid = threadIdx.x;
  float m = -3.0e38f;
  for (int j = tid; j < 65536; j += 256) m = fmaxf(m, base[j]);
  red[tid] = m;
  __syncthreads();
  for (int off = 128; off; off >>= 1) {
    if (tid < off) red[tid] = fmaxf(red[tid], red[tid + off]);
    __syncthreads();
  }
  float M = red[0];
  if (tid == 0) mruB[bid] = M;
  unsigned short* out = er + ((size_t)bid << 16);
  for (int j = tid; j < 65536; j += 256) {
    int s = j >> 8, t = j & 255;
    out[(t << 8) + s] = f2bf(__expf(base[j] - M));
  }
}

// ---------------------------------------------------------------------------
// Kernel A (per step w): one workgroup per valid span (b, n), n in [0, N-w).
// Computes mm via WMMA (K = w padded to 32), exact my = max(Y), and writes
// eY[b][n][t*256+s] = bf16( max(mm,TINY) * exp(ml[s]+mr[t]-my) ).
// ---------------------------------------------------------------------------
__global__ __launch_bounds__(256) void kA(const float* __restrict__ chart,
                                          unsigned short* __restrict__ eY,
                                          float* __restrict__ myB, int w) {
  const int N = 30, S = 256;
  int b = blockIdx.x, n = blockIdx.y;       // n < N - w (all valid)
  int tid = threadIdx.x;

  __shared__ unsigned short elT[256][36];   // [s][k] bf16 of exp(left - ml)
  __shared__ unsigned short eRT[256][36];   // [t][k] bf16 of exp(right - mr)
  __shared__ float mlS[256], mrS[256];
  __shared__ float redS[8];
  __shared__ float myS;

  // ---- phase 1: per-column maxima + exp tables (k = u-1, zero-padded) ----
  {
    int s = tid;
    const float* cb = chart + (size_t)b * N * N * S;
    float ml = NEGV, mr = NEGV;
    for (int u = 1; u <= w; ++u) {
      ml = fmaxf(ml, cb[((size_t)n * N + (u - 1)) * S + s]);
      mr = fmaxf(mr, cb[((size_t)(n + u) * N + (w - u)) * S + s]);
    }
    mlS[s] = ml; mrS[s] = mr;
    for (int u = 1; u <= w; ++u) {
      elT[s][u - 1] = f2bf(__expf(cb[((size_t)n * N + (u - 1)) * S + s] - ml));
      eRT[s][u - 1] = f2bf(__expf(cb[((size_t)(n + u) * N + (w - u)) * S + s] - mr));
    }
    for (int k = w; k < 32; ++k) { elT[s][k] = 0; eRT[s][k] = 0; }
  }
  __syncthreads();

  int wv = tid >> 5, lane = tid & 31;
  int lo16 = lane & 15, hi = lane >> 4;
  int shi = hi * 8;

  // ---- phase 2a: mm via WMMA, find my = max(ml[s]+mr[t]+log(max(mm,TINY)))
  float localMax = -3.0e38f;
  for (int ti = wv; ti < 256; ti += 8) {
    int s0 = (ti >> 4) << 4, t0 = (ti & 15) << 4;
    BF16Frag fa, fb;
    const unsigned short* ra = &elT[s0 + lo16][0];
    const unsigned short* rb = &eRT[t0 + lo16][0];
#pragma unroll
    for (int e = 0; e < 16; ++e) { fa.u[e] = ra[kmapA(e, hi)]; fb.u[e] = rb[kmapB(e, hi)]; }
    v8f acc = {0.f, 0.f, 0.f, 0.f, 0.f, 0.f, 0.f, 0.f};
    acc = __builtin_amdgcn_wmma_f32_16x16x32_bf16(false, fa.v, false, fb.v,
                                                  (short)0, acc, false, false);
    int tt = t0 + lo16;
#pragma unroll
    for (int r = 0; r < 8; ++r) {
      int ss = s0 + r + shi;
      float y = mlS[ss] + mrS[tt] + __logf(fmaxf(acc[r], TINYV));
      localMax = fmaxf(localMax, y);
    }
  }
  for (int off = 16; off; off >>= 1)
    localMax = fmaxf(localMax, __shfl_xor(localMax, off, 32));
  if (lane == 0) redS[wv] = localMax;
  __syncthreads();
  if (tid == 0) {
    float m = redS[0];
    for (int i = 1; i < 8; ++i) m = fmaxf(m, redS[i]);
    myS = m;
    myB[b * N + n] = m;
  }
  __syncthreads();
  float my = myS;

  // ---- phase 2b: recompute mm (cheap, K<=32) and emit eY in bf16 ----
  unsigned short* eyrow = eY + ((size_t)(b * N + n) << 16);
  for (int ti = wv; ti < 256; ti += 8) {
    int s0 = (ti >> 4) << 4, t0 = (ti & 15) << 4;
    BF16Frag fa, fb;
    const unsigned short* ra = &elT[s0 + lo16][0];
    const unsigned short* rb = &eRT[t0 + lo16][0];
#pragma unroll
    for (int e = 0; e < 16; ++e) { fa.u[e] = ra[kmapA(e, hi)]; fb.u[e] = rb[kmapB(e, hi)]; }
    v8f acc = {0.f, 0.f, 0.f, 0.f, 0.f, 0.f, 0.f, 0.f};
    acc = __builtin_amdgcn_wmma_f32_16x16x32_bf16(false, fa.v, false, fb.v,
                                                  (short)0, acc, false, false);
    int tt = t0 + lo16;
    unsigned short* dst = eyrow + ((size_t)tt << 8) + s0 + shi;  // 8 contiguous
#pragma unroll
    for (int r = 0; r < 8; ++r) {
      int ss = s0 + r + shi;
      float ev = fmaxf(acc[r], TINYV) * __expf(mlS[ss] + mrS[tt] - my);
      dst[r] = f2bf(ev);
    }
  }
}

// ---------------------------------------------------------------------------
// Kernel B (per step w): grid (b, a_tile 0..7, n_tile). M=16 span rows,
// N=16 nonterminals, K=65536 split across 8 waves (8192 each), then a
// deterministic fixed-order LDS sum of the 8 partials + log epilogue into the chart.
// ---------------------------------------------------------------------------
__global__ __launch_bounds__(256) void kB(const unsigned short* __restrict__ eY,
                                          const unsigned short* __restrict__ er,
                                          const float* __restrict__ myB,
                                          const float* __restrict__ mruB,
                                          float* __restrict__ chart, int w) {
  const int N = 30, S = 256, NT = 128;
  int b = blockIdx.x, a0 = blockIdx.y << 4, n0 = blockIdx.z << 4;
  int nValid = N - w;
  int tid = threadIdx.x, wv = tid >> 5, lane = tid & 31;
  int lo16 = lane & 15, hi = lane >> 4;

  __shared__ float part[8][256];

  int n = n0 + lo16;
  int nClamped = n < N ? n : (N - 1);                 // stay in-bounds
  const unsigned short* arow = eY + ((size_t)(b * N + nClamped) << 16);
  const unsigned short* brow = er + (((size_t)b * NT + (a0 + lo16)) << 16);
  bool aValid = (n < nValid);

  v8f acc = {0.f, 0.f, 0.f, 0.f, 0.f, 0.f, 0.f, 0.f};
  int kBase = wv << 13;                               // 8192 K per wave
  for (int kk = 0; kk < 8192; kk += 32) {
    int k0 = kBase + kk;
    BF16Frag fa, fb;
#pragma unroll
    for (int e = 0; e < 16; ++e) {
      unsigned short av = arow[k0 + kmapA(e, hi)];
      fa.u[e] = aValid ? av : (unsigned short)0;
      fb.u[e] = brow[k0 + kmapB(e, hi)];
    }
    acc = __builtin_amdgcn_wmma_f32_16x16x32_bf16(false, fa.v, false, fb.v,
                                                  (short)0, acc, false, false);
  }

  int shi = hi * 8;
#pragma unroll
  for (int r = 0; r < 8; ++r) part[wv][(r + shi) * 16 + lo16] = acc[r];
  __syncthreads();

  int i = tid, M = i >> 4, ac = i & 15;
  float sum = 0.f;
#pragma unroll
  for (int p = 0; p < 8; ++p) sum += part[p][i];       // fixed order: deterministic
  int nOut = n0 + M;
  if (nOut < nValid) {
    float Z = myB[b * N + nOut] + mruB[b * NT + a0 + ac] +
              __logf(fmaxf(sum, TINYV));
    chart[(((size_t)(b * N + nOut)) * N + w) * S + (a0 + ac)] = Z;
  }
}

// ---------------------------------------------------------------------------
// Root: out[b] = -logsumexp_a( chart[b,0,len-1,a] + roots[b,a] )
// ---------------------------------------------------------------------------
__global__ __launch_bounds__(128) void kRoot(const float* __restrict__ chart,
                                             const float* __restrict__ roots,
                                             const int* __restrict__ lengths,
                                             float* __restrict__ out) {
  const int N = 30, S = 256, NT = 128;
  int b = blockIdx.x, a = threadIdx.x;
  int len = lengths[b];
  float v = chart[(((size_t)(b * N)) * N + (len - 1)) * S + a] + roots[b * NT + a];
  __shared__ float sm[4], ss[4];
  __shared__ float bm;
  int wv = a >> 5, lane = a & 31;
  float m = v;
  for (int off = 16; off; off >>= 1) m = fmaxf(m, __shfl_xor(m, off, 32));
  if (lane == 0) sm[wv] = m;
  __syncthreads();
  if (a == 0) {
    float t = sm[0];
    for (int i = 1; i < 4; ++i) t = fmaxf(t, sm[i]);
    bm = t;
  }
  __syncthreads();
  float e = __expf(v - bm);
  for (int off = 16; off; off >>= 1) e += __shfl_xor(e, off, 32);
  if (lane == 0) ss[wv] = e;
  __syncthreads();
  if (a == 0) out[b] = -(bm + __logf(ss[0] + ss[1] + ss[2] + ss[3]));
}

// ---------------------------------------------------------------------------
extern "C" void kernel_launch(void* const* d_in, const int* in_sizes, int n_in,
                              void* d_out, int out_size, void* d_ws, size_t ws_size,
                              hipStream_t stream) {
  (void)in_sizes; (void)n_in; (void)out_size; (void)ws_size;
  const int B = 16, N = 30, NT = 128, S = 256;
  const float* terms   = (const float*)d_in[0];
  const float* rules   = (const float*)d_in[1];
  const float* roots   = (const float*)d_in[2];
  const int*   lengths = (const int*)d_in[3];
  float* out = (float*)d_out;

  // workspace carve-up (~346 MB total):
  //   chart  : B*N*N*S f32            (14.7 MB)
  //   my     : B*N f32, mru: B*NT f32 (tiny)
  //   eY     : B*N*S*S bf16           (62.9 MB)
  //   er     : B*NT*S*S bf16          (268.4 MB)
  char* ws = (char*)d_ws;
  size_t off = 0;
  float* chart = (float*)(ws + off); off += (size_t)B * N * N * S * 4;
  float* myB   = (float*)(ws + off); off += (size_t)B * N * 4;
  float* mruB  = (float*)(ws + off); off += (size_t)B * NT * 4;
  off = (off + 255) & ~(size_t)255;
  unsigned short* eYB = (unsigned short*)(ws + off); off += (size_t)B * N * S * S * 2;
  unsigned short* erB = (unsigned short*)(ws + off); off += (size_t)B * NT * S * S * 2;

  kInit<<<(B * N * N * S + 255) / 256, 256, 0, stream>>>(chart, terms);
  kRules<<<B * NT, 256, 0, stream>>>(rules, erB, mruB);

  for (int w = 1; w < N; ++w) {
    kA<<<dim3(B, N - w), 256, 0, stream>>>(chart, eYB, myB, w);
    int mt = (N - w + 15) / 16;
    kB<<<dim3(B, 8, mt), 256, 0, stream>>>(eYB, erB, myB, mruB, chart, w);
  }

  kRoot<<<B, 128, 0, stream>>>(chart, roots, lengths, out);
}
